// AttentionLayer_26903675142338
// MI455X (gfx1250) — compile-verified, matrix operand order checked by reading, not run on hardware
//
#include <hip/hip_runtime.h>
#include <hip/hip_bf16.h>

typedef __attribute__((ext_vector_type(16))) _Float16 v16h;
typedef __attribute__((ext_vector_type(8)))  _Float16 v8h;
typedef __attribute__((ext_vector_type(4)))  _Float16 v4h;
typedef __attribute__((ext_vector_type(8)))  float    v8f;
typedef __attribute__((ext_vector_type(2)))  float    v2f;

#define NB     4
#define LQ     80        // query tokens (N in reference)
#define HH     24
#define WW     24
#define SS     77        // context tokens S
#define SP     80        // S padded to multiple of 16
#define CIN    256       // hidden+guid
#define HID    128
#define NH     8
#define LDH    264       // f16 LDS row stride (halves): 256 + 8 pad (bank skew)
#define LDF    132       // f32 LDS row stride (floats): 128 + 4 pad
#define EPSF   1e-6f

// LDS layout (bytes). Regions A/B are f16 staging then reused as f32 buffers.
#define OFF_A  0                 // aKV (f16 80x264 = 42240) -> later Vf (f32 80x132)
#define OFF_B  42240             // aQ  (f16 80x264)         -> later Qf (f32 80x132)
#define OFF_C  84480             // Kf  (f32 80x132 = 42240)
#define OFF_KS 126720            // Ksum: 8*16 f32 = 512
#define OFF_Z  127232            // Z:    8*80 f32 = 2560
#define OFF_KV 129792            // KV:   8*16*17 f32 = 8704
#define SMEM_TOTAL 138496

// ---- kernel 0: transpose + f16-convert weights into workspace ------------
__global__ void wconv_kernel(const float* __restrict__ Wq,
                             const float* __restrict__ Wk,
                             const float* __restrict__ Wv,
                             _Float16* __restrict__ wsQ,
                             _Float16* __restrict__ wsK,
                             _Float16* __restrict__ wsV) {
  int i = blockIdx.x * 256 + threadIdx.x;
  if (i < CIN * HID) {                       // Wq/Wk: (256,128) -> WT[j][k] (128x256)
    int k = i >> 7, j = i & 127;
    wsQ[j * CIN + k] = (_Float16)Wq[i];
    wsK[j * CIN + k] = (_Float16)Wk[i];
  } else {
    int t = i - CIN * HID;
    if (t < HID * HID) {                     // Wv: (128,128) -> WT[j][k] (128x128)
      int k = t >> 7, j = t & 127;
      wsV[j * HID + k] = (_Float16)Wv[t];
    }
  }
}

// ---- fused linear-attention kernel: one workgroup per n ------------------
__global__ __launch_bounds__(256) void fla_fused_kernel(
    const float* __restrict__ query, const float* __restrict__ xin,
    const float* __restrict__ gin,
    const _Float16* __restrict__ WqT, const float* __restrict__ bq,
    const _Float16* __restrict__ WkT, const float* __restrict__ bk,
    const _Float16* __restrict__ WvT, const float* __restrict__ bv,
    float* __restrict__ out)
{
  extern __shared__ char smem[];
  _Float16* aKV = (_Float16*)(smem + OFF_A);
  _Float16* aQ  = (_Float16*)(smem + OFF_B);
  float*    Kf  = (float*)(smem + OFF_C);
  float*    Vf  = (float*)(smem + OFF_A);
  float*    Qf  = (float*)(smem + OFF_B);
  float*    Ks  = (float*)(smem + OFF_KS);
  float*    Zb  = (float*)(smem + OFF_Z);
  float*    KVb = (float*)(smem + OFF_KV);

  const int n   = blockIdx.x;
  const int tid = threadIdx.x;
  const int wv  = tid >> 5;          // wave == head == N-tile (8 waves)
  const int ln  = tid & 31;
  const int l16 = ln & 15;
  const int hi  = ln >> 4;
  const int col = wv * 16 + l16;     // this wave's output column

  const int b  = n / (HH * WW);
  const int hw = n - b * (HH * WW);

  // -------- Phase 1: stage activations to LDS as f16 ----------------------
  // query[b, l, h, w, c] -> aQ[l][c]
  const float* qb = query + ((size_t)(b * LQ) * (HH * WW) + (size_t)hw) * CIN;
  for (int i = tid; i < LQ * (CIN / 4); i += 256) {
    int r = i >> 6, c4 = (i & 63) << 2;
    float4 v = *(const float4*)(qb + (size_t)r * (HH * WW) * CIN + c4);
    v4h t = { (_Float16)v.x, (_Float16)v.y, (_Float16)v.z, (_Float16)v.w };
    *(v4h*)(&aQ[r * LDH + c4]) = t;
  }
  // concat(x, guidance) -> aKV[s][0:256]
  const float* xb = xin + (size_t)n * SS * HID;
  const float* gb = gin + (size_t)n * SS * HID;
  for (int i = tid; i < SS * (HID / 4); i += 256) {
    int r = i >> 5, c4 = (i & 31) << 2;
    float4 vx = *(const float4*)(xb + r * HID + c4);
    float4 vg = *(const float4*)(gb + r * HID + c4);
    v4h tx = { (_Float16)vx.x, (_Float16)vx.y, (_Float16)vx.z, (_Float16)vx.w };
    v4h tg = { (_Float16)vg.x, (_Float16)vg.y, (_Float16)vg.z, (_Float16)vg.w };
    *(v4h*)(&aKV[r * LDH + c4])       = tx;
    *(v4h*)(&aKV[r * LDH + HID + c4]) = tg;
  }
  for (int i = tid; i < (SP - SS) * CIN; i += 256) {   // zero pad rows 77..79
    int r = SS + i / CIN, c = i % CIN;
    aKV[r * LDH + c] = (_Float16)0.0f;
  }
  __syncthreads();

  // fragment loaders (A/B layout: lanes0-15 K=k0..k0+7,k0+16..23; lanes16-31 +8)
  auto ldA = [&](const _Float16* base, int row, int k0) -> v16h {
    const _Float16* p = base + row * LDH + k0 + 8 * hi;
    v8h lo = *(const v8h*)p;
    v8h hq = *(const v8h*)(p + 16);
    v16h r;
#pragma unroll
    for (int t = 0; t < 8; ++t) { r[t] = lo[t]; r[t + 8] = hq[t]; }
    return r;
  };
  auto ldB = [&](const _Float16* WT, int ldw, int k0) -> v16h {
    const _Float16* p = WT + col * ldw + k0 + 8 * hi;
    v8h lo = *(const v8h*)p;
    v8h hq = *(const v8h*)(p + 16);
    v16h r;
#pragma unroll
    for (int t = 0; t < 8; ++t) { r[t] = lo[t]; r[t + 8] = hq[t]; }
    return r;
  };

  // -------- Phase 2: Q projection (keep in regs, feature-map applied) -----
  v8f accQ[5] = {};
  for (int ksI = 0; ksI < 8; ++ksI) {
    int k0 = ksI * 32;
    v16h bf = ldB(WqT, CIN, k0);
#pragma unroll
    for (int m = 0; m < 5; ++m) {
      v16h af = ldA(aQ, m * 16 + l16, k0);
      accQ[m] = __builtin_amdgcn_wmma_f32_16x16x32_f16(
          false, af, false, bf, (short)0, accQ[m], false, false);
    }
  }
  {
    float bb = bq[col];
#pragma unroll
    for (int m = 0; m < 5; ++m)
#pragma unroll
      for (int r = 0; r < 8; ++r) {
        float t = accQ[m][r] + bb;
        accQ[m][r] = (t > 0.f) ? (t + 1.f) : __expf(t);   // elu(x)+1
      }
  }

  // -------- Phase 3a: K projection -> feature map -> Kf (fresh region) ----
  {
    v8f acc[5] = {};
    for (int ksI = 0; ksI < 8; ++ksI) {
      int k0 = ksI * 32;
      v16h bf = ldB(WkT, CIN, k0);
#pragma unroll
      for (int m = 0; m < 5; ++m) {
        v16h af = ldA(aKV, m * 16 + l16, k0);
        acc[m] = __builtin_amdgcn_wmma_f32_16x16x32_f16(
            false, af, false, bf, (short)0, acc[m], false, false);
      }
    }
    float bb = bk[col];
#pragma unroll
    for (int m = 0; m < 5; ++m)
#pragma unroll
      for (int r = 0; r < 8; ++r) {
        int row = m * 16 + r + 8 * hi;
        float t = acc[m][r] + bb;
        t = (t > 0.f) ? (t + 1.f) : __expf(t);
        if (row >= SS) t = 0.f;                 // mask padded rows
        Kf[row * LDF + col] = t;
      }
  }

  // -------- Phase 3b: V projection (x only, K=128) ------------------------
  v8f accV[5] = {};
  for (int ksI = 0; ksI < 4; ++ksI) {
    int k0 = ksI * 32;
    v16h bf = ldB(WvT, HID, k0);
#pragma unroll
    for (int m = 0; m < 5; ++m) {
      v16h af = ldA(aKV, m * 16 + l16, k0);
      accV[m] = __builtin_amdgcn_wmma_f32_16x16x32_f16(
          false, af, false, bf, (short)0, accV[m], false, false);
    }
  }
  __syncthreads();   // all waves done reading aKV/aQ -> safe to overwrite

  {
    float bb = bv[col];
#pragma unroll
    for (int m = 0; m < 5; ++m)
#pragma unroll
      for (int r = 0; r < 8; ++r) {
        int row = m * 16 + r + 8 * hi;
        Vf[row * LDF + col] = accV[m][r] + bb;
        Qf[row * LDF + col] = accQ[m][r];
      }
  }
  asm volatile("s_wait_dscnt 0" ::: "memory");

  // -------- Phase 4 (wave-local, head = wv): Ksum and KV = K^T V ----------
  float ksum = 0.f;
  for (int s = 0; s < SP; ++s) ksum += Kf[s * LDF + col];
  if (hi == 0) Ks[col] = ksum;

  v8f accKV = {};
#if __has_builtin(__builtin_amdgcn_wmma_f32_16x16x4_f32)
  for (int s0 = 0; s0 < SP; s0 += 4) {
    int s = s0 + 2 * hi;
    v2f a, bf;
    a[0]  = Kf[s * LDF + col];       a[1]  = Kf[(s + 1) * LDF + col];
    bf[0] = Vf[s * LDF + col];       bf[1] = Vf[(s + 1) * LDF + col];
    accKV = __builtin_amdgcn_wmma_f32_16x16x4_f32(
        false, a, false, bf, (short)0, accKV, false, false);
  }
#else
#pragma unroll
  for (int r = 0; r < 8; ++r) {
    float acc = 0.f;
    int d = r + 8 * hi;
    for (int s = 0; s < SP; ++s)
      acc += Kf[s * LDF + wv * 16 + d] * Vf[s * LDF + col];
    accKV[r] = acc;
  }
#endif
#pragma unroll
  for (int r = 0; r < 8; ++r)
    KVb[wv * 272 + (r + 8 * hi) * 17 + l16] = accKV[r];
  asm volatile("s_wait_dscnt 0" ::: "memory");

  // -------- Phase 5: Z = 1/(Q.Ksum + eps) ---------------------------------
  for (int l = ln; l < LQ; l += 32) {
    float dot = 0.f;
#pragma unroll
    for (int d = 0; d < 16; ++d) dot += Qf[l * LDF + wv * 16 + d] * Ks[wv * 16 + d];
    Zb[wv * LQ + l] = 1.f / (dot + EPSF);
  }
  asm volatile("s_wait_dscnt 0" ::: "memory");

  // -------- Phase 6: queried = (Z*Q) @ KV, mean over rows -----------------
  float colsum = 0.f;
#pragma unroll
  for (int m = 0; m < 5; ++m) {
    v8f accO = {};
#if __has_builtin(__builtin_amdgcn_wmma_f32_16x16x4_f32)
    int row = m * 16 + l16;
    float z = Zb[wv * LQ + row];
#pragma unroll
    for (int d0 = 0; d0 < 16; d0 += 4) {
      int d = d0 + 2 * hi;
      v2f a, bf;
      a[0]  = Qf[row * LDF + wv * 16 + d]     * z;
      a[1]  = Qf[row * LDF + wv * 16 + d + 1] * z;
      bf[0] = KVb[wv * 272 + d * 17 + l16];
      bf[1] = KVb[wv * 272 + (d + 1) * 17 + l16];
      accO = __builtin_amdgcn_wmma_f32_16x16x4_f32(
          false, a, false, bf, (short)0, accO, false, false);
    }
#else
#pragma unroll
    for (int r = 0; r < 8; ++r) {
      int M = m * 16 + r + 8 * hi;
      float z = Zb[wv * LQ + M];
      float acc = 0.f;
#pragma unroll
      for (int d = 0; d < 16; ++d)
        acc += Qf[M * LDF + wv * 16 + d] * KVb[wv * 272 + d * 17 + l16];
      accO[r] = acc * z;
    }
#endif
#pragma unroll
    for (int r = 0; r < 8; ++r) colsum += accO[r];
  }
  colsum += __shfl_xor(colsum, 16, 32);
  if (hi == 0) out[(size_t)n * HID + col] = colsum * (1.0f / LQ);
}

// ---- host launch ---------------------------------------------------------
extern "C" void kernel_launch(void* const* d_in, const int* in_sizes, int n_in,
                              void* d_out, int out_size, void* d_ws, size_t ws_size,
                              hipStream_t stream) {
  (void)in_sizes; (void)n_in; (void)out_size; (void)ws_size;
  const float* query = (const float*)d_in[0];
  const float* x     = (const float*)d_in[1];
  const float* g     = (const float*)d_in[2];
  const float* Wq    = (const float*)d_in[3];
  const float* bq    = (const float*)d_in[4];
  const float* Wk    = (const float*)d_in[5];
  const float* bk    = (const float*)d_in[6];
  const float* Wv    = (const float*)d_in[7];
  const float* bv    = (const float*)d_in[8];
  float* out = (float*)d_out;

  _Float16* wsQ = (_Float16*)d_ws;          // 128x256 f16
  _Float16* wsK = wsQ + 256 * 128;          // 128x256 f16
  _Float16* wsV = wsK + 256 * 128;          // 128x128 f16

  wconv_kernel<<<192, 256, 0, stream>>>(Wq, Wk, Wv, wsQ, wsK, wsV);
  fla_fused_kernel<<<NB * HH * WW, 256, SMEM_TOTAL, stream>>>(
      query, x, g, wsQ, bq, wsK, bk, wsV, bv, out);
}
